// LocalFrameAttention_46806553592109
// MI455X (gfx1250) — compile-verified, matrix operand order checked by reading, not run on hardware
//
#include <hip/hip_runtime.h>
#include <hip/hip_bf16.h>

typedef __attribute__((ext_vector_type(16))) _Float16 v16h;
typedef __attribute__((ext_vector_type(8)))  float    v8f;

union Frag16 {                 // 16x32 f16 A/B fragment (8 VGPRs)
    v16h     v;
    _Float16 h[16];
};
union FragC {                  // 16x16 f32 C/D fragment (8 VGPRs)
    v8f   v;
    float f[8];
};

// VGPR j (0..7) of an A/B fragment holds the K-pair starting at:
//   K = 2*(j&3) + 16*(j>>2) + 8*half      (half = lane>>4)
__device__ __forceinline__ int kpair(int j, int half) {
    return 2 * (j & 3) + 16 * (j >> 2) + 8 * half;
}

__device__ __forceinline__ void loadpair(const float* p, _Float16* dst) {
    float2 t = *(const float2*)p;          // 8B aligned (even K index)
    dst[0] = (_Float16)t.x;
    dst[1] = (_Float16)t.y;
}
__device__ __forceinline__ void loadpair(const _Float16* p, _Float16* dst) {
    dst[0] = p[0];
    dst[1] = p[1];
}

// --- CDNA5 async global->LDS copy (16B per lane), ASYNCcnt-tracked ----------
// Inline asm per CDNA5 ISA / CDNA5 ISA:
//   GLOBAL_LOAD_ASYNC_TO_LDS_B128  vdst(LDS byte addr), vaddr(64b global), off
__device__ __forceinline__ void async_copy16(const void* g, unsigned lds_off) {
    unsigned long long ga = (unsigned long long)g;
    asm volatile("global_load_async_to_lds_b128 %0, %1, off"
                 :: "v"(lds_off), "v"(ga) : "memory");
}

__device__ __forceinline__ void wait_async0() {
#if __has_builtin(__builtin_amdgcn_s_wait_asynccnt)
    __builtin_amdgcn_s_wait_asynccnt(0);
#else
    asm volatile("s_wait_asynccnt 0x0" ::: "memory");
#endif
}

// ---------------------------------------------------------------------------
// Wave-tiled WMMA GEMM:  C[M,N] = A[M,K] * W[N,K]^T
// Each wave computes a 32x64 tile of C (2x4 grid of 16x16 WMMA tiles).
// A: f32 or f16 (converted on the fly), W: f32, C: f16 or f32.
// ---------------------------------------------------------------------------
template <typename AT, typename OT>
__global__ void wmma_gemm_kernel(const AT* __restrict__ A,
                                 const float* __restrict__ W,
                                 OT* __restrict__ C,
                                 int M, int N, int K) {
    const int lane = threadIdx.x & 31;
    const int wave = threadIdx.x >> 5;
    const int r    = lane & 15;
    const int half = lane >> 4;

    const int  mtiles = M >> 5;            // 32-row tiles
    const int  ntiles = N >> 6;            // 64-col tiles
    const long tid    = (long)blockIdx.x * 8 + wave;
    if (tid >= (long)mtiles * ntiles) return;   // wave-uniform

    const int mt = (int)(tid % mtiles);
    const int nt = (int)(tid / mtiles);
    const int m0 = mt << 5;
    const int n0 = nt << 6;

    FragC acc[2][4];
#pragma unroll
    for (int mi = 0; mi < 2; ++mi)
#pragma unroll
        for (int ni = 0; ni < 4; ++ni)
#pragma unroll
            for (int i = 0; i < 8; ++i) acc[mi][ni].f[i] = 0.0f;

    for (int k0 = 0; k0 < K; k0 += 32) {
        Frag16 a[2];
#pragma unroll
        for (int mi = 0; mi < 2; ++mi) {
            const AT* arow = A + (long)(m0 + 16 * mi + r) * K + k0;
            __builtin_prefetch(arow + 32, 0, 0);
#pragma unroll
            for (int j = 0; j < 8; ++j) {
                const int kk = kpair(j, half);
                loadpair(arow + kk, &a[mi].h[2 * j]);
            }
        }
        Frag16 b[4];
#pragma unroll
        for (int ni = 0; ni < 4; ++ni) {
            const float* wrow = W + (long)(n0 + 16 * ni + r) * K + k0;
            __builtin_prefetch(wrow + 32, 0, 0);
#pragma unroll
            for (int j = 0; j < 8; ++j) {
                const int kk = kpair(j, half);
                loadpair(wrow + kk, &b[ni].h[2 * j]);
            }
        }
#pragma unroll
        for (int mi = 0; mi < 2; ++mi)
#pragma unroll
            for (int ni = 0; ni < 4; ++ni)
                acc[mi][ni].v = __builtin_amdgcn_wmma_f32_16x16x32_f16(
                    false, a[mi].v, false, b[ni].v,
                    (short)0, acc[mi][ni].v, false, false);
    }

#pragma unroll
    for (int mi = 0; mi < 2; ++mi)
#pragma unroll
        for (int ni = 0; ni < 4; ++ni)
#pragma unroll
            for (int i = 0; i < 8; ++i) {
                const int row = m0 + 16 * mi + i + 8 * half;
                const int col = n0 + 16 * ni + r;
                C[(long)row * N + col] = (OT)acc[mi][ni].f[i];
            }
}

// ---------------------------------------------------------------------------
// Flash attention over chunk windows.
// qkv: [8192, 3072] f16  (cols 0..1023 = Q, 1024..2047 = K, 2048..3071 = V,
//                         col layout within each = h*64 + d)
// attn: [8192, 1024] f16 output (col = h*64 + d)
// grid: b(2) * c(4) * h(16) * qtile(8) = 1024 blocks, 256 threads (8 waves),
// each wave owns 16 query rows. All waves in a block share one K/V stream,
// which is staged into LDS once per 32-key block via async-to-LDS copies.
// ---------------------------------------------------------------------------
__global__ void flash_attn_kernel(const _Float16* __restrict__ qkv,
                                  _Float16* __restrict__ attn) {
    constexpr int E  = 3072;   // qkv row stride
    constexpr int AO = 1024;   // attn row stride
    __shared__ _Float16 kt[32 * 64];      // staged K tile [key][d]   (4 KB)
    __shared__ _Float16 vt[32 * 64];      // staged V tile [key][d]   (4 KB)
    __shared__ _Float16 pb[8][16 * 32];   // per-wave P staging tile  (8 KB)

    const int tid  = threadIdx.x;
    const int lane = tid & 31;
    const int wave = tid >> 5;
    const int r    = lane & 15;
    const int half = lane >> 4;

    const int bid = blockIdx.x;
    const int qt  = bid & 7;
    const int h   = (bid >> 3) & 15;
    const int c   = (bid >> 7) & 3;
    const int b   = bid >> 9;

    const long tq_base = (long)(b * 4 + c) * 1024;
    const long tk_base = (long)(b * 4 + (c > 0 ? c - 1 : 0)) * 1024;
    const int  Tk      = (c > 0) ? 2048 : 1024;
    const int  qrow0   = qt * 128 + wave * 16;   // within chunk

    // Q fragments (two K=32 halves of HD=64), pre-scaled by 1/sqrt(HD)
    Frag16 qa[2];
    {
        const _Float16* qrow = qkv + (tq_base + qrow0 + r) * E + h * 64;
#pragma unroll
        for (int kq = 0; kq < 2; ++kq)
#pragma unroll
            for (int j = 0; j < 8; ++j) {
                const int kk = 32 * kq + kpair(j, half);
                qa[kq].h[2 * j]     = (_Float16)((float)qrow[kk] * 0.125f);
                qa[kq].h[2 * j + 1] = (_Float16)((float)qrow[kk + 1] * 0.125f);
            }
    }

    FragC o[4];
#pragma unroll
    for (int jt = 0; jt < 4; ++jt)
#pragma unroll
        for (int i = 0; i < 8; ++i) o[jt].f[i] = 0.0f;
    float m_run[8], l_run[8];
#pragma unroll
    for (int i = 0; i < 8; ++i) { m_run[i] = -3.0e38f; l_run[i] = 0.0f; }

    for (int k0 = 0; k0 < Tk; k0 += 32) {
        // Cooperative async stage of the 32x64 K and V tiles (16B per thread)
        {
            const int row = tid >> 3;     // 0..31 key rows
            const int ch  = tid & 7;      // 8 x 16B chunks per row
            const _Float16* gk =
                qkv + (tk_base + k0 + row) * E + 1024 + h * 64 + ch * 8;
            async_copy16(gk, (unsigned)(unsigned long long)&kt[row * 64 + ch * 8]);
            async_copy16(gk + 1024,
                         (unsigned)(unsigned long long)&vt[row * 64 + ch * 8]);
        }
        wait_async0();
        __syncthreads();

        // K fragments from LDS: B[kk=d][n=key]
        Frag16 kb[2][2];
#pragma unroll
        for (int t = 0; t < 2; ++t)
#pragma unroll
            for (int kq = 0; kq < 2; ++kq)
#pragma unroll
                for (int j = 0; j < 8; ++j) {
                    const int kk  = 32 * kq + kpair(j, half);
                    const int idx = (16 * t + r) * 64 + kk;
                    kb[t][kq].h[2 * j]     = kt[idx];
                    kb[t][kq].h[2 * j + 1] = kt[idx + 1];
                }

        // S = Q * K^T  (16x32 of logits as two 16x16 tiles)
        FragC s[2];
#pragma unroll
        for (int t = 0; t < 2; ++t) {
#pragma unroll
            for (int i = 0; i < 8; ++i) s[t].f[i] = 0.0f;
            s[t].v = __builtin_amdgcn_wmma_f32_16x16x32_f16(
                false, qa[0].v, false, kb[t][0].v, (short)0, s[t].v, false, false);
            s[t].v = __builtin_amdgcn_wmma_f32_16x16x32_f16(
                false, qa[1].v, false, kb[t][1].v, (short)0, s[t].v, false, false);
        }

        // Online softmax (per C-VGPR row; 16-lane butterflies stay in-half)
        float alpha[8];
#pragma unroll
        for (int i = 0; i < 8; ++i) {
            float rm = fmaxf(s[0].f[i], s[1].f[i]);
#pragma unroll
            for (int off = 1; off < 16; off <<= 1)
                rm = fmaxf(rm, __shfl_xor(rm, off, 32));
            const float mn = fmaxf(m_run[i], rm);
            const float p0 = __expf(s[0].f[i] - mn);
            const float p1 = __expf(s[1].f[i] - mn);
            const float a  = __expf(m_run[i] - mn);
            float rs = p0 + p1;
#pragma unroll
            for (int off = 1; off < 16; off <<= 1)
                rs += __shfl_xor(rs, off, 32);
            l_run[i] = l_run[i] * a + rs;
            m_run[i] = mn;
            alpha[i] = a;
            s[0].f[i] = p0;
            s[1].f[i] = p1;
        }
#pragma unroll
        for (int jt = 0; jt < 4; ++jt)
#pragma unroll
            for (int i = 0; i < 8; ++i) o[jt].f[i] *= alpha[i];

        // Repack P (C layout) -> A layout via per-wave LDS tile
#pragma unroll
        for (int t = 0; t < 2; ++t)
#pragma unroll
            for (int i = 0; i < 8; ++i)
                pb[wave][(i + 8 * half) * 32 + 16 * t + r] = (_Float16)s[t].f[i];
        __syncthreads();
        Frag16 pa;
#pragma unroll
        for (int j = 0; j < 8; ++j) {
            const int kk = kpair(j, half);
            pa.h[2 * j]     = pb[wave][r * 32 + kk];
            pa.h[2 * j + 1] = pb[wave][r * 32 + kk + 1];
        }

        // O += P * V  (four 16x16 HD tiles); V fragments from LDS:
        // B[kk=key][n=d] -> transpose gather is LDS-local now
#pragma unroll
        for (int jt = 0; jt < 4; ++jt) {
            Frag16 vb;
#pragma unroll
            for (int j = 0; j < 8; ++j) {
                const int kk = kpair(j, half);
                vb.h[2 * j]     = vt[kk * 64 + 16 * jt + r];
                vb.h[2 * j + 1] = vt[(kk + 1) * 64 + 16 * jt + r];
            }
            o[jt].v = __builtin_amdgcn_wmma_f32_16x16x32_f16(
                false, pa.v, false, vb.v, (short)0, o[jt].v, false, false);
        }

        __syncthreads();   // protect kt/vt/pb before next iteration's stage
    }

    // Normalize and store
#pragma unroll
    for (int jt = 0; jt < 4; ++jt)
#pragma unroll
        for (int i = 0; i < 8; ++i) {
            const long t = tq_base + qrow0 + i + 8 * half;
            attn[t * AO + h * 64 + 16 * jt + r] =
                (_Float16)(o[jt].f[i] / l_run[i]);
        }
}

// ---------------------------------------------------------------------------
extern "C" void kernel_launch(void* const* d_in, const int* in_sizes, int n_in,
                              void* d_out, int out_size, void* d_ws, size_t ws_size,
                              hipStream_t stream) {
    const float* x     = (const float*)d_in[0];   // [2,16,256,1024] = [8192,1024]
    const float* w_qkv = (const float*)d_in[1];   // [3072,1024]
    const float* w_out = (const float*)d_in[2];   // [1024,1024]
    float*       out   = (float*)d_out;           // [8192,1024]

    constexpr int  M = 8192;    // total tokens
    constexpr int  D = 1024;
    constexpr int  E = 3072;

    _Float16* qkv_ws  = (_Float16*)d_ws;                       // 8192*3072 f16
    _Float16* attn_ws = qkv_ws + (size_t)M * E;                // 8192*1024 f16

    // 1) QKV projection: qkv = x * w_qkv^T   (f32 -> f16 out)
    {
        const int waves  = (M / 32) * (E / 64);   // 12288
        const int blocks = (waves + 7) / 8;       // 1536
        wmma_gemm_kernel<float, _Float16>
            <<<blocks, 256, 0, stream>>>(x, w_qkv, qkv_ws, M, E, D);
    }

    // 2) Windowed flash attention (async-LDS staged K/V)
    flash_attn_kernel<<<1024, 256, 0, stream>>>(qkv_ws, attn_ws);

    // 3) Output projection: out = attn * w_out^T   (f16 in, f32 out)
    {
        const int waves  = (M / 32) * (D / 64);   // 4096
        const int blocks = (waves + 7) / 8;       // 512
        wmma_gemm_kernel<_Float16, float>
            <<<blocks, 256, 0, stream>>>(attn_ws, w_out, out, M, D, D);
    }
}
